// GridAttention_37134287241920
// MI455X (gfx1250) — matrix operand order repaired and checked by reading, not compiled, not run on hardware
//
#include <hip/hip_runtime.h>
#include <hip/hip_bf16.h>

typedef __bf16 bf16;
typedef __attribute__((ext_vector_type(16))) __bf16 v16bf;
typedef __attribute__((ext_vector_type(8)))  float  v8f;

#define NHEADS 8
#define HD     64
#define S_TOK  3072
#define D_EMB  512
#define GH     64
#define GW     48

// ---------------- WMMA fragment helpers (CDNA5 wave32 layouts) ----------------

// A-matrix 16x32 bf16: lane<16 -> K=0..15 ; lane>=16 -> K=16..31
static __device__ __forceinline__ v16bf load_a_frag(const bf16* base, int ld) {
  int lane = threadIdx.x & 31;
  int row  = lane & 15;
  int kb   = (lane >> 4) << 4;          // 0 or 16
  const bf16* p = base + row * ld + kb;
  v16bf a;
#pragma unroll
  for (int e = 0; e < 16; ++e) a[e] = p[e];
  return a;
}

// B-fragment from K-contiguous storage: matrix stored [N][K] row-major (i.e. B^T),
// lane<16 -> K=0..15, lane>=16 -> K=16..31; 32B contiguous per lane (2x ds_load_b128)
static __device__ __forceinline__ v16bf load_bt_frag(const bf16* base, int ld) {
  int lane = threadIdx.x & 31;
  int n    = lane & 15;
  int kb   = (lane >> 4) << 4;          // 0 or 16
  const bf16* p = base + n * ld + kb;
  v16bf b;
#pragma unroll
  for (int e = 0; e < 16; ++e) b[e] = p[e];
  return b;
}

static __device__ __forceinline__ v8f wmma_bf16(v16bf a, v16bf b, v8f c) {
  return __builtin_amdgcn_wmma_f32_16x16x32_bf16(false, a, false, b, (short)0, c, false, false);
}

// ---------------- Kernel 1: weight fp32 -> bf16, transposed to [N][K] ----------------

__global__ void convert_weights(const float* __restrict__ wqkv, const float* __restrict__ wout,
                                bf16* __restrict__ wqkvT, bf16* __restrict__ woutT) {
  const int n1 = 3 * D_EMB * D_EMB;   // 1536 x 512
  const int n2 = D_EMB * D_EMB;       //  512 x 512
  for (int i = blockIdx.x * blockDim.x + threadIdx.x; i < n1 + n2; i += gridDim.x * blockDim.x) {
    if (i < n1) {
      int nn = i >> 9, kk = i & 511;
      wqkvT[i] = (bf16)wqkv[kk * (3 * D_EMB) + nn];
    } else {
      int j = i - n1;
      int nn = j >> 9, kk = j & 511;
      woutT[j] = (bf16)wout[kk * D_EMB + nn];
    }
  }
}

// ---------------- Kernels 2 & 4: double-buffered WMMA GEMM ----------------
// MODE 0: A = x fp32 [6144,512], W^T = w_qkv^T bf16 [1536,512], scatter->Q/K/V (scale in Q)
// MODE 1: A = O bf16 [6144,512], W^T = w_out^T bf16 [512,512], write fp32 out
template <int MODE>
__global__ void __launch_bounds__(256) gemm_kernel(const void* __restrict__ Aptr,
                                                   const bf16* __restrict__ WbT,
                                                   bf16* __restrict__ Qb, bf16* __restrict__ Kb,
                                                   bf16* __restrict__ Vb, float* __restrict__ Out) {
  constexpr int K = D_EMB;
  __shared__ __attribute__((aligned(16))) bf16 sA[2][64 * 40];    // padded stride 40
  __shared__ __attribute__((aligned(16))) bf16 sBt[2][128 * 40];  // W^T tile, K-contiguous

  const int m0 = blockIdx.x * 64;
  const int n0 = blockIdx.y * 128;
  const int t = threadIdx.x;
  const int lane = t & 31, wv = t >> 5;
  const int msub = (wv & 3) * 16;
  const int nsub = (wv >> 2) * 64;
  const int arow = t >> 2, acol = (t & 3) * 8;   // A tile 64x32
  const int brow = t >> 1, bcol = (t & 1) * 16;  // B^T tile 128x32

  float4 fa0, fa1; int4 ia; int4 wb0, wb1;
  auto loadA = [&](int k0) {
    if constexpr (MODE == 0) {
      const float* src = (const float*)Aptr + (size_t)(m0 + arow) * K + k0 + acol;
      fa0 = *(const float4*)src; fa1 = *(const float4*)(src + 4);
    } else {
      ia = *(const int4*)((const bf16*)Aptr + (size_t)(m0 + arow) * K + k0 + acol);
    }
  };
  auto loadB = [&](int k0) {
    const int4* src = (const int4*)(WbT + (size_t)(n0 + brow) * K + k0 + bcol);
    wb0 = src[0]; wb1 = src[1];
  };
  auto stage = [&](int bufi) {
    bf16* da = &sA[bufi][arow * 40 + acol];
    if constexpr (MODE == 0) {
      da[0] = (bf16)fa0.x; da[1] = (bf16)fa0.y; da[2] = (bf16)fa0.z; da[3] = (bf16)fa0.w;
      da[4] = (bf16)fa1.x; da[5] = (bf16)fa1.y; da[6] = (bf16)fa1.z; da[7] = (bf16)fa1.w;
    } else {
      *(int4*)da = ia;
    }
    int4* db = (int4*)&sBt[bufi][brow * 40 + bcol];
    db[0] = wb0; db[1] = wb1;
  };

  v8f zero = {0.f, 0.f, 0.f, 0.f, 0.f, 0.f, 0.f, 0.f};
  v8f acc[4] = {zero, zero, zero, zero};

  loadA(0); loadB(0);
  int buf = 0;
  for (int ks = 0; ks < 16; ++ks) {
    stage(buf);
    __syncthreads();
    if (ks < 15) { loadA((ks + 1) * 32); loadB((ks + 1) * 32); }  // overlap with WMMAs
    v16bf a = load_a_frag(&sA[buf][msub * 40], 40);
#pragma unroll
    for (int nt = 0; nt < 4; ++nt) {
      v16bf b = load_bt_frag(&sBt[buf][(nsub + nt * 16) * 40], 40);
      acc[nt] = wmma_bf16(a, b, acc[nt]);
    }
    buf ^= 1;
  }

#pragma unroll
  for (int nt = 0; nt < 4; ++nt) {
#pragma unroll
    for (int r = 0; r < 8; ++r) {
      int mg = m0 + msub + r + ((lane >> 4) << 3);
      int ng = n0 + nsub + nt * 16 + (lane & 15);
      float v = acc[nt][r];
      if constexpr (MODE == 0) {
        int bidx = mg / S_TOK;
        int s    = mg - bidx * S_TOK;
        int which = ng >> 9;          // 0=q 1=k 2=v
        int rem   = ng & 511;
        int head  = rem >> 6;
        int hd    = rem & 63;
        size_t idx = (((size_t)(bidx * NHEADS + head)) * S_TOK + s) * HD + hd;
        if (which == 0)      Qb[idx] = (bf16)(v * 0.125f);   // fold 1/sqrt(64) into Q
        else if (which == 1) Kb[idx] = (bf16)v;
        else                 Vb[idx] = (bf16)v;
      } else {
        Out[(size_t)mg * D_EMB + ng] = v;
      }
    }
  }
}

// ---------------- Kernel 3: flash attention w/ 2D relative bias ----------------
// grid = 16 (b,head) * 48 q-tiles ; block = 128 threads = 4 waves, each wave owns 16 queries.
// Double-buffered K / V^T tiles in LDS; V stored transposed so PV B-frags are K-contiguous.
__global__ void __launch_bounds__(128) attention_kernel(const bf16* __restrict__ Qb,
                                                        const bf16* __restrict__ Kb,
                                                        const bf16* __restrict__ Vb,
                                                        const float* __restrict__ rowtab,
                                                        const float* __restrict__ coltab,
                                                        bf16* __restrict__ Ob) {
  __shared__ __attribute__((aligned(16))) bf16 sK[2][64 * 72];    // [token][hd], padded
  __shared__ __attribute__((aligned(16))) bf16 sVt[2][64 * 72];   // [hd][token], padded
  __shared__ __attribute__((aligned(16))) bf16 sP[4][16 * 64];    // wave-private P staging
  __shared__ float srow[2 * GH - 1];
  __shared__ float scol[2 * GW - 1];

  const int bh = blockIdx.x / 48;
  const int qt = blockIdx.x % 48;
  const int b = bh >> 3, head = bh & 7;
  const int t = threadIdx.x, lane = t & 31, wv = t >> 5;
  const int kcol = lane & 15;
  const int q0 = qt * 64;
  const int vr = t & 63, vp = t >> 6;   // staging: token vr, hd range vp*32..+31

  if (t < 2 * GH - 1) srow[t] = rowtab[t * NHEADS + head];
  if (t < 2 * GW - 1) scol[t] = coltab[t * NHEADS + head];

  // Q fragments held in registers for the whole key loop (hd 0-31 and 32-63)
  const bf16* Qp = Qb + ((size_t)bh * S_TOK + q0 + wv * 16) * HD;
  v16bf aq0 = load_a_frag(Qp, HD);
  v16bf aq1 = load_a_frag(Qp + 32, HD);

  v8f zero = {0.f, 0.f, 0.f, 0.f, 0.f, 0.f, 0.f, 0.f};
  v8f ov[4] = {zero, zero, zero, zero};
  float mx[8], ls[8];
  int myq[8], myrow[8], mycol[8];
#pragma unroll
  for (int r = 0; r < 8; ++r) {
    mx[r] = -3.0e38f; ls[r] = 0.f;
    int q = q0 + wv * 16 + r + ((lane >> 4) << 3);
    myq[r] = q; myrow[r] = q / GW; mycol[r] = q - myrow[r] * GW;
  }

  const bf16* Kbase = Kb + (size_t)bh * S_TOK * HD;
  const bf16* Vbase = Vb + (size_t)bh * S_TOK * HD;

  // register prefetch of tile 0
  int4 rk[4];
  union { int4 q[4]; bf16 h[32]; } rv;
  {
    const int4* gk = (const int4*)(Kbase + (size_t)vr * HD + vp * 32);
    const int4* gv = (const int4*)(Vbase + (size_t)vr * HD + vp * 32);
#pragma unroll
    for (int i = 0; i < 4; ++i) { rk[i] = gk[i]; rv.q[i] = gv[i]; }
  }

  int buf = 0;
  for (int kt = 0; kt < 48; ++kt) {
    { // regs -> LDS: K straight (int4), V transposed (b16 scatter)
      int4* dk = (int4*)&sK[buf][vr * 72 + vp * 32];
#pragma unroll
      for (int i = 0; i < 4; ++i) dk[i] = rk[i];
#pragma unroll
      for (int j = 0; j < 32; ++j) sVt[buf][(vp * 32 + j) * 72 + vr] = rv.h[j];
    }
    __syncthreads();

    if (kt < 47) { // issue next tile's global loads; WMMAs below hide the latency
      const int4* gk = (const int4*)(Kbase + (size_t)((kt + 1) * 64 + vr) * HD + vp * 32);
      const int4* gv = (const int4*)(Vbase + (size_t)((kt + 1) * 64 + vr) * HD + vp * 32);
#pragma unroll
      for (int i = 0; i < 4; ++i) { rk[i] = gk[i]; rv.q[i] = gv[i]; }
      if (kt < 46) { // L2 prefetch two tiles ahead (128 threads x 64B = full 8KB tile)
        __builtin_prefetch(Kbase + (size_t)((kt + 2) * 64) * HD + t * 32, 0, 3);
        __builtin_prefetch(Vbase + (size_t)((kt + 2) * 64) * HD + t * 32, 0, 3);
      }
    }

    // S = Q * K^T  (16 queries x 64 keys), 8 WMMAs, contiguous B-frags from sK
    v8f sc[4] = {zero, zero, zero, zero};
#pragma unroll
    for (int kk = 0; kk < 2; ++kk) {
      v16bf a = (kk == 0) ? aq0 : aq1;
#pragma unroll
      for (int nt = 0; nt < 4; ++nt) {
        v16bf bf = load_bt_frag(&sK[buf][(nt * 16) * 72 + kk * 32], 72);
        sc[nt] = wmma_bf16(a, bf, sc[nt]);
      }
    }

    // + relative position bias (per-head rows cached in LDS)
#pragma unroll
    for (int nt = 0; nt < 4; ++nt) {
      int ks = kt * 64 + nt * 16 + kcol;
      int krow = ks / GW;
      int kc = ks - krow * GW;
#pragma unroll
      for (int r = 0; r < 8; ++r) {
        int dr = myrow[r] - krow + (GH - 1);   // always in [0,126]
        int dc = mycol[r] - kc + (GW - 1);     // always in [0,94]
        sc[nt][r] = sc[nt][r] + srow[dr] + scol[dc];
      }
    }

    // online softmax: rows live across 16-lane groups, reduce with shfl_xor
#pragma unroll
    for (int r = 0; r < 8; ++r) {
      float v = fmaxf(fmaxf(sc[0][r], sc[1][r]), fmaxf(sc[2][r], sc[3][r]));
#pragma unroll
      for (int off = 1; off < 16; off <<= 1) v = fmaxf(v, __shfl_xor(v, off, 32));
      float mnew = fmaxf(mx[r], v);
      float alpha = __expf(mx[r] - mnew);
      mx[r] = mnew;
      float rs = 0.f;
#pragma unroll
      for (int nt = 0; nt < 4; ++nt) {
        float p = __expf(sc[nt][r] - mnew);
        sc[nt][r] = p;
        rs += p;
      }
#pragma unroll
      for (int off = 1; off < 16; off <<= 1) rs += __shfl_xor(rs, off, 32);
      ls[r] = ls[r] * alpha + rs;
#pragma unroll
      for (int nt = 0; nt < 4; ++nt) ov[nt][r] = ov[nt][r] * alpha;
    }

    // stage P (C-layout) into wave-private LDS to reload in A-layout
    bf16* pw = sP[wv];
#pragma unroll
    for (int nt = 0; nt < 4; ++nt) {
#pragma unroll
      for (int r = 0; r < 8; ++r) {
        int row = r + ((lane >> 4) << 3);
        pw[row * 64 + nt * 16 + kcol] = (bf16)sc[nt][r];
      }
    }

    // O += P * V, 8 WMMAs; B-frags K-contiguous from transposed V
#pragma unroll
    for (int kk = 0; kk < 2; ++kk) {
      v16bf a = load_a_frag(pw + kk * 32, 64);
#pragma unroll
      for (int nt = 0; nt < 4; ++nt) {
        v16bf bf = load_bt_frag(&sVt[buf][(nt * 16) * 72 + kk * 32], 72);
        ov[nt] = wmma_bf16(a, bf, ov[nt]);
      }
    }
    buf ^= 1;
  }

  // normalize and write O as [b, s, head*64+hd] bf16
#pragma unroll
  for (int r = 0; r < 8; ++r) {
    float inv = 1.f / ls[r];
    size_t obase = ((size_t)b * S_TOK + myq[r]) * D_EMB + head * HD;
#pragma unroll
    for (int nt = 0; nt < 4; ++nt)
      Ob[obase + nt * 16 + kcol] = (bf16)(ov[nt][r] * inv);
  }
}

// ---------------- launcher ----------------

extern "C" void kernel_launch(void* const* d_in, const int* in_sizes, int n_in,
                              void* d_out, int out_size, void* d_ws, size_t ws_size,
                              hipStream_t stream) {
  const float* x       = (const float*)d_in[0];
  const float* w_qkv   = (const float*)d_in[1];
  const float* w_out   = (const float*)d_in[2];
  const float* rel_row = (const float*)d_in[3];
  const float* rel_col = (const float*)d_in[4];
  float* out = (float*)d_out;

  char* ws = (char*)d_ws;
  bf16* wqkvT = (bf16*)(ws);              // [1536][512] bf16 = 1,572,864 B
  bf16* woutT = (bf16*)(ws + 1572864);    //  [512][512] bf16 =   524,288 B
  bf16* Qb    = (bf16*)(ws + 2097152);    // 2*8*3072*64*2 = 6,291,456 B each
  bf16* Kb    = (bf16*)(ws + 8388608);
  bf16* Vb    = (bf16*)(ws + 14680064);
  bf16* Ob    = (bf16*)(ws + 20971520);   // 6144*512*2, total ws use ~27.3 MB

  convert_weights<<<512, 256, 0, stream>>>(w_qkv, w_out, wqkvT, woutT);
  gemm_kernel<0><<<dim3(96, 12), 256, 0, stream>>>(x, wqkvT, Qb, Kb, Vb, nullptr);
  attention_kernel<<<16 * 48, 128, 0, stream>>>(Qb, Kb, Vb, rel_row, rel_col, Ob);
  gemm_kernel<1><<<dim3(96, 4), 256, 0, stream>>>(Ob, woutT, nullptr, nullptr, nullptr, out);
}
